// GPT2_70317204570643
// MI455X (gfx1250) — compile-verified, matrix operand order checked by reading, not run on hardware
//
#include <hip/hip_runtime.h>
#include <hip/hip_bf16.h>

typedef __bf16 bf16;
typedef unsigned int u32;
typedef unsigned long long u64;
typedef __attribute__((ext_vector_type(16))) __bf16 v16bf;
typedef __attribute__((ext_vector_type(8)))  __bf16 v8bf;
typedef __attribute__((ext_vector_type(8)))  float  v8f;
typedef __attribute__((ext_vector_type(4)))  u32    v4u;
typedef __attribute__((ext_vector_type(8)))  u32    v8u;

#define TSEQ 2048
#define DMODEL 768
#define NHEAD 12
#define HDIM 64
#define NLAYER 6
#define NBATCH 2
#define NROWS (NBATCH * TSEQ)   /* 4096 */
#define VOC 50257
#define VPAD 50304               /* VOC rounded up to 128 for aligned B-tile loads */

static __device__ __forceinline__ v16bf cat8(v8bf lo, v8bf hi) {
  return __builtin_shufflevector(lo, hi, 0,1,2,3,4,5,6,7,8,9,10,11,12,13,14,15);
}

// ---------------------------------------------------------------------------
// TDM: issue one tensor_load_to_lds moving a 2-D bf16 tile (tile_w x tile_h)
// from global (row stride = stride elems) into LDS at lds_off. Descriptor per
// CDNA5 ISA ch.8 (D# groups); groups 2/3 zero (2-D: tile_dim2..4 unused).
// Must be executed by exactly one wave; completion via TENSORcnt.
// ---------------------------------------------------------------------------
static __device__ __forceinline__ void tdm_load_2d_bf16(
    u32 lds_off, const void* gsrc, u32 tile_w, u32 tile_h,
    u32 tensor_w, u32 tensor_h, u32 stride) {
  u64 ga = (u64)gsrc;
  v4u g0;
  g0[0] = 1u;                                    // count=1, user descriptor
  g0[1] = lds_off;                               // lds_addr (bytes)
  g0[2] = (u32)(ga & 0xFFFFFFFFu);               // global_addr[31:0]
  g0[3] = (u32)((ga >> 32) & 0x01FFFFFFu) | (2u << 30);  // addr[56:32] | type=2
  v8u g1;
  g1[0] = (1u << 16);                            // wg_mask=0 | data_size=1 (2B)
  g1[1] = (tensor_w & 0xFFFFu) << 16;            // tensor_dim0[15:0]
  g1[2] = (tensor_w >> 16) | ((tensor_h & 0xFFFFu) << 16);
  g1[3] = (tensor_h >> 16) | (tile_w << 16);     // tensor_dim1 hi | tile_dim0
  g1[4] = tile_h & 0xFFFFu;                      // tile_dim1 | tile_dim2=0
  g1[5] = stride;                                // tensor_dim0_stride[31:0]
  g1[6] = 0u;                                    // stride hi | dim1_stride lo
  g1[7] = 0u;
  v4u gz = {0u, 0u, 0u, 0u};
  asm volatile("tensor_load_to_lds %0, %1, %2, %3"
               :: "s"(g0), "s"(g1), "s"(gz), "s"(gz)
               : "memory");
}

// ---------------------------------------------------------------------------
// fp32 -> bf16 elementwise convert (weights)
// ---------------------------------------------------------------------------
__global__ __launch_bounds__(256) void gpt2_f2bf(const float* __restrict__ s,
                                                 bf16* __restrict__ d, size_t n) {
  size_t i = (size_t)blockIdx.x * 256 + threadIdx.x;
  size_t stride = (size_t)gridDim.x * 256;
  for (; i < n; i += stride) d[i] = (bf16)s[i];
}

// wte [V, D] fp32 -> wteT [D, VPAD] bf16 (pad cols left as-is; never stored out)
__global__ __launch_bounds__(256) void gpt2_wteT(const float* __restrict__ s,
                                                 bf16* __restrict__ d) {
  size_t i = (size_t)blockIdx.x * 256 + threadIdx.x;
  if (i >= (size_t)VOC * DMODEL) return;
  int v  = (int)(i / DMODEL);
  int dd = (int)(i % DMODEL);
  d[(size_t)dd * VPAD + v] = (bf16)s[i];
}

// ---------------------------------------------------------------------------
// Embedding: x[row,:] = wte[idx[row],:] + wpe[row % T,:]
// ---------------------------------------------------------------------------
__global__ __launch_bounds__(256) void gpt2_embed(const int* __restrict__ idx,
                                                  const float* __restrict__ wte,
                                                  const float* __restrict__ wpe,
                                                  float* __restrict__ x) {
  int row = blockIdx.x;
  int t = row & (TSEQ - 1);
  int tok = idx[row];
  for (int c = threadIdx.x; c < DMODEL; c += 256)
    x[(size_t)row * DMODEL + c] =
        wte[(size_t)tok * DMODEL + c] + wpe[(size_t)t * DMODEL + c];
}

// ---------------------------------------------------------------------------
// LayerNorm over D=768, fp32 in -> bf16 out
// ---------------------------------------------------------------------------
static __device__ __forceinline__ float gpt2_block_reduce(float v, float* sh) {
  __syncthreads();
#pragma unroll
  for (int m = 16; m >= 1; m >>= 1) v += __shfl_xor(v, m, 32);
  if ((threadIdx.x & 31) == 0) sh[threadIdx.x >> 5] = v;
  __syncthreads();
  if (threadIdx.x == 0) {
    float t = 0.f;
#pragma unroll
    for (int i = 0; i < 8; ++i) t += sh[i];
    sh[8] = t;
  }
  __syncthreads();
  return sh[8];
}

__global__ __launch_bounds__(256) void gpt2_layernorm(const float* __restrict__ x,
                                                      const float* __restrict__ g,
                                                      const float* __restrict__ b,
                                                      bf16* __restrict__ out) {
  __shared__ float sh[9];
  const int row = blockIdx.x;
  const float* xr = x + (size_t)row * DMODEL;
  float v[3];
#pragma unroll
  for (int i = 0; i < 3; ++i) v[i] = xr[threadIdx.x + i * 256];
  float s = v[0] + v[1] + v[2];
  s = gpt2_block_reduce(s, sh);
  float mu = s * (1.f / DMODEL);
  float q = 0.f;
#pragma unroll
  for (int i = 0; i < 3; ++i) { float d = v[i] - mu; q += d * d; }
  q = gpt2_block_reduce(q, sh);
  float rstd = rsqrtf(q * (1.f / DMODEL) + 1e-5f);
#pragma unroll
  for (int i = 0; i < 3; ++i) {
    int c = threadIdx.x + i * 256;
    out[(size_t)row * DMODEL + c] = (bf16)((v[i] - mu) * rstd * g[c] + b[c]);
  }
}

// ---------------------------------------------------------------------------
// WMMA bf16 GEMM:  C[M,N] = A[M,K](bf16) @ B[K,N](bf16)  (+bias, gelu, +res)
// Block: 256 thr (8 waves), tile 128x128, K-step 32.
// A tile staged by TDM (tensor_load_to_lds, wave 0, TENSORcnt);
// B tile staged transposed via VGPRs. FLAGS: 1=bias 2=gelu 4=res 8=bf16 out
// ---------------------------------------------------------------------------
template <int FLAGS>
__global__ __launch_bounds__(256) void gpt2_gemm(
    const bf16* __restrict__ A, const bf16* __restrict__ B,
    const float* __restrict__ bias, const float* __restrict__ res,
    float* __restrict__ Cf, bf16* __restrict__ Cb,
    int N, int K, int ldb, int ldc) {
  __shared__ alignas(16) bf16 sA[128 * 32];   // [m][k]
  __shared__ alignas(16) bf16 sB[128 * 32];   // transposed: [n][k]
  const int tid = threadIdx.x;
  const int lane = tid & 31;
  const int w = tid >> 5;
  const int lm = lane & 15, kh = lane >> 4;
  const int wr = w >> 1, wc = w & 1;       // 4 wave-rows x 2 wave-cols
  const int bM = blockIdx.y * 128;
  const int bN = blockIdx.x * 128;

  v8f acc[2][4];
#pragma unroll
  for (int mi = 0; mi < 2; ++mi)
#pragma unroll
    for (int ni = 0; ni < 4; ++ni) {
      v8f z = {0.f, 0.f, 0.f, 0.f, 0.f, 0.f, 0.f, 0.f};
      acc[mi][ni] = z;
    }

  for (int k0 = 0; k0 < K; k0 += 32) {
    if (w == 0) {  // TDM: A tile 128 rows x 32 cols, one DMA op
      tdm_load_2d_bf16((u32)(u64)&sA[0], &A[(size_t)bM * K + k0],
                       /*tile_w=*/32, /*tile_h=*/128,
                       /*tensor_w=*/(u32)K, /*tensor_h=*/(u32)NROWS,
                       /*stride=*/(u32)K);
    }
    if (k0 + 32 < K) {  // prefetch next tiles -> global_prefetch_b8
      __builtin_prefetch(&A[(size_t)(bM + (tid >> 1)) * K + k0 + 32], 0, 1);
      __builtin_prefetch(&B[(size_t)(k0 + 32 + (tid >> 3)) * ldb + bN + (tid & 7) * 16], 0, 1);
    }
    {  // stage B tile 32x128 -> transposed LDS [n][k]
      int kr = tid >> 3, c0 = (tid & 7) * 16;
      const bf16* src = &B[(size_t)(k0 + kr) * ldb + bN + c0];
      alignas(16) bf16 tmp[16];
      *(uint4*)&tmp[0] = *(const uint4*)(src);
      *(uint4*)&tmp[8] = *(const uint4*)(src + 8);
#pragma unroll
      for (int i = 0; i < 16; ++i) sB[(c0 + i) * 32 + kr] = tmp[i];
    }
    if (w == 0) __builtin_amdgcn_s_wait_tensorcnt(0);
    __syncthreads();

    v16bf af[2], bfrag[4];
#pragma unroll
    for (int mi = 0; mi < 2; ++mi) {
      const bf16* p = &sA[(wr * 32 + mi * 16 + lm) * 32 + kh * 8];
      af[mi] = cat8(*(const v8bf*)p, *(const v8bf*)(p + 16));
    }
#pragma unroll
    for (int ni = 0; ni < 4; ++ni) {
      const bf16* p = &sB[(wc * 64 + ni * 16 + lm) * 32 + kh * 8];
      bfrag[ni] = cat8(*(const v8bf*)p, *(const v8bf*)(p + 16));
    }
#pragma unroll
    for (int mi = 0; mi < 2; ++mi)
#pragma unroll
      for (int ni = 0; ni < 4; ++ni)
        acc[mi][ni] = __builtin_amdgcn_wmma_f32_16x16x32_bf16(
            false, af[mi], false, bfrag[ni], (short)0, acc[mi][ni], false, false);
    __syncthreads();
  }

#pragma unroll
  for (int mi = 0; mi < 2; ++mi)
#pragma unroll
    for (int ni = 0; ni < 4; ++ni)
#pragma unroll
      for (int r = 0; r < 8; ++r) {
        int gm = bM + wr * 32 + mi * 16 + r + kh * 8;
        int gn = bN + wc * 64 + ni * 16 + lm;
        if (gn < N) {
          float v = acc[mi][ni][r];
          if (FLAGS & 1) v += bias[gn];
          if (FLAGS & 2) v = 0.5f * v * (1.f + erff(v * 0.70710678118f));
          if (FLAGS & 4) v += res[(size_t)gm * ldc + gn];
          if (FLAGS & 8) Cb[(size_t)gm * ldc + gn] = (bf16)v;
          else           Cf[(size_t)gm * ldc + gn] = v;
        }
      }
}

// ---------------------------------------------------------------------------
// Flash attention (causal), 1 wave per (b, h, 16-row q tile).
// K tile staged with GLOBAL_LOAD_ASYNC_TO_LDS_B128 (ASYNCcnt); S = Q@K^T and
// O += P@V through v_wmma_f32_16x16x32_bf16; online softmax in C-tile layout.
// ---------------------------------------------------------------------------
__global__ __launch_bounds__(32) void gpt2_attention(const bf16* __restrict__ Q,
                                                     const bf16* __restrict__ Kc,
                                                     const bf16* __restrict__ Vc,
                                                     bf16* __restrict__ O) {
  __shared__ alignas(16) bf16 ldsK[32 * 64];   // [key][hd]
  __shared__ alignas(16) bf16 ldsVt[64 * 32];  // [hd][key]
  __shared__ alignas(16) bf16 ldsP[16 * 32];   // [m][key]

  const int bid = blockIdx.x;
  const int qt = bid & 127;                    // T/16 = 128 q tiles
  const int h  = (bid >> 7) % NHEAD;
  const int b  = bid / (128 * NHEAD);
  const int qbase = qt * 16;
  const int rowbase = b * TSEQ + qbase;
  const int colbase = h * HDIM;
  const int lane = threadIdx.x;
  const int lm = lane & 15, kh = lane >> 4;

  // Q fragments (2 k-steps over HD=64), straight from global
  v16bf qf[2];
  {
    const bf16* qp = &Q[(size_t)(rowbase + lm) * DMODEL + colbase];
#pragma unroll
    for (int s = 0; s < 2; ++s) {
      const bf16* p = qp + s * 32 + kh * 8;
      qf[s] = cat8(*(const v8bf*)p, *(const v8bf*)(p + 16));
    }
  }

  v8f acc[4];
#pragma unroll
  for (int t = 0; t < 4; ++t) {
    v8f z = {0.f, 0.f, 0.f, 0.f, 0.f, 0.f, 0.f, 0.f};
    acc[t] = z;
  }
  float mrow[8], lrow[8];
#pragma unroll
  for (int r = 0; r < 8; ++r) { mrow[r] = -1e30f; lrow[r] = 0.f; }

  const u32 ldsKbase = (u32)(u64)&ldsK[0];
  const int nkb = (qbase >> 5) + 1;  // 32-key blocks covering keys <= qbase+15
  for (int kb = 0; kb < nkb; ++kb) {
    const int keybase = kb * 32;
    const int grow = b * TSEQ + keybase;
    {  // stage K [32 keys][64] via async global->LDS copies (lane = key row)
      const bf16* ks = &Kc[(size_t)(grow + lane) * DMODEL + colbase];
      u32 ldst = ldsKbase + (u32)(lane * 128);
#pragma unroll
      for (int j = 0; j < 64; j += 8) {
        asm volatile("global_load_async_to_lds_b128 %0, %1, off"
                     :: "v"(ldst + (u32)(j * 2)), "v"((u64)(ks + j))
                     : "memory");
      }
      // stage V transposed [hd][key] via VGPRs
      const bf16* vs = &Vc[(size_t)(grow + lane) * DMODEL + colbase];
#pragma unroll
      for (int j0 = 0; j0 < 64; j0 += 8) {
        alignas(16) bf16 tmp[8];
        *(uint4*)tmp = *(const uint4*)&vs[j0];
#pragma unroll
        for (int j = 0; j < 8; ++j) ldsVt[(j0 + j) * 32 + lane] = tmp[j];
      }
      asm volatile("s_wait_asynccnt 0" ::: "memory");
    }
    __syncthreads();  // single-wave WG: LDS pipe in-order

    // S tiles: keys [keybase, keybase+16) and [keybase+16, keybase+32)
    v8f s0 = {0.f, 0.f, 0.f, 0.f, 0.f, 0.f, 0.f, 0.f};
    v8f s1 = s0;
#pragma unroll
    for (int s = 0; s < 2; ++s) {
      const bf16* p0 = &ldsK[lm * 64 + s * 32 + kh * 8];
      v16bf kb0 = cat8(*(const v8bf*)p0, *(const v8bf*)(p0 + 16));
      const bf16* p1 = &ldsK[(16 + lm) * 64 + s * 32 + kh * 8];
      v16bf kb1 = cat8(*(const v8bf*)p1, *(const v8bf*)(p1 + 16));
      s0 = __builtin_amdgcn_wmma_f32_16x16x32_bf16(false, qf[s], false, kb0,
                                                   (short)0, s0, false, false);
      s1 = __builtin_amdgcn_wmma_f32_16x16x32_bf16(false, qf[s], false, kb1,
                                                   (short)0, s1, false, false);
    }

    // online softmax; row M = r + 8*kh lives across 16 lanes of a half-wave
#pragma unroll
    for (int r = 0; r < 8; ++r) {
      const int qrow = qbase + r + kh * 8;
      float a0 = s0[r] * 0.125f;  // HD^-0.5
      float a1 = s1[r] * 0.125f;
      if (keybase + lm > qrow) a0 = -1e30f;
      if (keybase + 16 + lm > qrow) a1 = -1e30f;
      float mx = fmaxf(a0, a1);
#pragma unroll
      for (int m = 8; m >= 1; m >>= 1) mx = fmaxf(mx, __shfl_xor(mx, m, 32));
      const float mnew = fmaxf(mrow[r], mx);
      const float p0 = __expf(a0 - mnew);
      const float p1 = __expf(a1 - mnew);
      float ps = p0 + p1;
#pragma unroll
      for (int m = 8; m >= 1; m >>= 1) ps += __shfl_xor(ps, m, 32);
      const float f = __expf(mrow[r] - mnew);
      lrow[r] = lrow[r] * f + ps;
      mrow[r] = mnew;
#pragma unroll
      for (int t = 0; t < 4; ++t) acc[t][r] *= f;
      const int prow = r + kh * 8;
      ldsP[prow * 32 + lm] = (bf16)p0;
      ldsP[prow * 32 + 16 + lm] = (bf16)p1;
    }

    // P (16x32) @ V (32x64): reload P as A fragment, Vt columns as B fragments
    const bf16* pp = &ldsP[lm * 32 + kh * 8];
    v16bf pf = cat8(*(const v8bf*)pp, *(const v8bf*)(pp + 16));
#pragma unroll
    for (int t = 0; t < 4; ++t) {
      const bf16* vp = &ldsVt[(t * 16 + lm) * 32 + kh * 8];
      v16bf vb = cat8(*(const v8bf*)vp, *(const v8bf*)(vp + 16));
      acc[t] = __builtin_amdgcn_wmma_f32_16x16x32_bf16(false, pf, false, vb,
                                                       (short)0, acc[t], false, false);
    }
    __syncthreads();
  }

#pragma unroll
  for (int t = 0; t < 4; ++t)
#pragma unroll
    for (int r = 0; r < 8; ++r) {
      size_t off = (size_t)(rowbase + r + kh * 8) * DMODEL + colbase + t * 16 + lm;
      O[off] = (bf16)(acc[t][r] / lrow[r]);
    }
}

// ---------------------------------------------------------------------------
// Host orchestration
// ---------------------------------------------------------------------------
extern "C" void kernel_launch(void* const* d_in, const int* in_sizes, int n_in,
                              void* d_out, int out_size, void* d_ws, size_t ws_size,
                              hipStream_t stream) {
  (void)in_sizes; (void)n_in; (void)out_size; (void)ws_size;
  const int*   idx  = (const int*)  d_in[0];
  const float* wte  = (const float*)d_in[1];
  const float* wpe  = (const float*)d_in[2];
  const float* Wq   = (const float*)d_in[3];
  const float* Wk   = (const float*)d_in[4];
  const float* Wv   = (const float*)d_in[5];
  const float* Wo   = (const float*)d_in[6];
  const float* ln1g = (const float*)d_in[7];
  const float* ln1b = (const float*)d_in[8];
  const float* ln2g = (const float*)d_in[9];
  const float* ln2b = (const float*)d_in[10];
  const float* W1   = (const float*)d_in[11];
  const float* b1   = (const float*)d_in[12];
  const float* W2   = (const float*)d_in[13];
  const float* b2   = (const float*)d_in[14];
  const float* lnfg = (const float*)d_in[15];
  const float* lnfb = (const float*)d_in[16];
  float* out = (float*)d_out;

  char* ws = (char*)d_ws;
  size_t off = 0;
  auto alloc = [&](size_t bytes) -> char* {
    char* p = ws + off;
    off += (bytes + 255) & ~(size_t)255;
    return p;
  };
  float* x   = (float*)alloc((size_t)NROWS * DMODEL * 4);
  bf16* h_bf = (bf16*) alloc((size_t)NROWS * DMODEL * 2);
  bf16* m_bf = (bf16*) alloc((size_t)NROWS * 4 * DMODEL * 2);
  bf16* q_bf = (bf16*) alloc((size_t)NROWS * DMODEL * 2);
  bf16* k_bf = (bf16*) alloc((size_t)NROWS * DMODEL * 2);
  bf16* v_bf = (bf16*) alloc((size_t)NROWS * DMODEL * 2);
  bf16* a_bf = (bf16*) alloc((size_t)NROWS * DMODEL * 2);
  bf16* wq_bf = (bf16*)alloc((size_t)NLAYER * DMODEL * DMODEL * 2);
  bf16* wk_bf = (bf16*)alloc((size_t)NLAYER * DMODEL * DMODEL * 2);
  bf16* wv_bf = (bf16*)alloc((size_t)NLAYER * DMODEL * DMODEL * 2);
  bf16* wo_bf = (bf16*)alloc((size_t)NLAYER * DMODEL * DMODEL * 2);
  bf16* w1_bf = (bf16*)alloc((size_t)NLAYER * DMODEL * 4 * DMODEL * 2);
  bf16* w2_bf = (bf16*)alloc((size_t)NLAYER * DMODEL * 4 * DMODEL * 2);
  bf16* wt_bf = (bf16*)alloc((size_t)DMODEL * VPAD * 2);

  const size_t nqkv = (size_t)NLAYER * DMODEL * DMODEL;
  const size_t nmlp = (size_t)NLAYER * DMODEL * 4 * DMODEL;
  gpt2_f2bf<<<4096, 256, 0, stream>>>(Wq, wq_bf, nqkv);
  gpt2_f2bf<<<4096, 256, 0, stream>>>(Wk, wk_bf, nqkv);
  gpt2_f2bf<<<4096, 256, 0, stream>>>(Wv, wv_bf, nqkv);
  gpt2_f2bf<<<4096, 256, 0, stream>>>(Wo, wo_bf, nqkv);
  gpt2_f2bf<<<8192, 256, 0, stream>>>(W1, w1_bf, nmlp);
  gpt2_f2bf<<<8192, 256, 0, stream>>>(W2, w2_bf, nmlp);
  {
    size_t nt = (size_t)VOC * DMODEL;
    gpt2_wteT<<<(unsigned)((nt + 255) / 256), 256, 0, stream>>>(wte, wt_bf);
  }
  gpt2_embed<<<NROWS, 256, 0, stream>>>(idx, wte, wpe, x);

  dim3 blk(256);
  dim3 gD(DMODEL / 128, NROWS / 128);          // 6 x 32
  dim3 g4D(4 * DMODEL / 128, NROWS / 128);     // 24 x 32
  dim3 gV(VPAD / 128, NROWS / 128);            // 393 x 32
  const int nattn = NBATCH * NHEAD * (TSEQ / 16);

  for (int l = 0; l < NLAYER; ++l) {
    const size_t oD  = (size_t)l * DMODEL * DMODEL;
    const size_t oM  = (size_t)l * DMODEL * 4 * DMODEL;
    gpt2_layernorm<<<NROWS, 256, 0, stream>>>(x, ln1g + l * DMODEL,
                                              ln1b + l * DMODEL, h_bf);
    gpt2_gemm<8><<<gD, blk, 0, stream>>>(h_bf, wq_bf + oD, nullptr, nullptr,
                                         nullptr, q_bf, DMODEL, DMODEL, DMODEL, DMODEL);
    gpt2_gemm<8><<<gD, blk, 0, stream>>>(h_bf, wk_bf + oD, nullptr, nullptr,
                                         nullptr, k_bf, DMODEL, DMODEL, DMODEL, DMODEL);
    gpt2_gemm<8><<<gD, blk, 0, stream>>>(h_bf, wv_bf + oD, nullptr, nullptr,
                                         nullptr, v_bf, DMODEL, DMODEL, DMODEL, DMODEL);
    gpt2_attention<<<nattn, 32, 0, stream>>>(q_bf, k_bf, v_bf, a_bf);
    gpt2_gemm<4><<<gD, blk, 0, stream>>>(a_bf, wo_bf + oD, nullptr, x,
                                         x, nullptr, DMODEL, DMODEL, DMODEL, DMODEL);
    gpt2_layernorm<<<NROWS, 256, 0, stream>>>(x, ln2g + l * DMODEL,
                                              ln2b + l * DMODEL, h_bf);
    gpt2_gemm<1 | 2 | 8><<<g4D, blk, 0, stream>>>(h_bf, w1_bf + oM,
                                                  b1 + (size_t)l * 4 * DMODEL, nullptr,
                                                  nullptr, m_bf,
                                                  4 * DMODEL, DMODEL, 4 * DMODEL, 4 * DMODEL);
    gpt2_gemm<1 | 4><<<gD, blk, 0, stream>>>(m_bf, w2_bf + oM,
                                             b2 + (size_t)l * DMODEL, x,
                                             x, nullptr, DMODEL, 4 * DMODEL, DMODEL, DMODEL);
  }
  gpt2_layernorm<<<NROWS, 256, 0, stream>>>(x, lnfg, lnfb, h_bf);
  gpt2_gemm<0><<<gV, blk, 0, stream>>>(h_bf, wt_bf, nullptr, nullptr,
                                       out, nullptr, VOC, DMODEL, VPAD, VOC);
}
